// EEGSMILESFuseMoE_19189913879367
// MI455X (gfx1250) — matrix operand order repaired and checked by reading, not compiled, not run on hardware
//
#include <hip/hip_runtime.h>
#include <hip/hip_bf16.h>
#include <math.h>

// ---------------------------------------------------------------------------
// CDNA5 (gfx1250) wave32 WMMA types
// ---------------------------------------------------------------------------
typedef __attribute__((ext_vector_type(16))) _Float16 v16h;
typedef __attribute__((ext_vector_type(8)))  _Float16 h8;
typedef __attribute__((ext_vector_type(8)))  float    v8f;

union Frag16 { v16h v; h8 h[2]; };
union HPack  { _Float16 h[2]; unsigned int u; };

#define LDS_STRIDE 40   // halves; 80B row pitch -> 16B-aligned b128 reads, spread banks

// ---------------------------------------------------------------------------
// Batched GEMM: C[e] = act(A[e] @ B[e] + bias[e])
//   REQUIRES: M % 16 == 0, K % 32 == 0, N % 128 == 0 (true for all launches)
//   A: MxK (row-major, lda even), B: KxN (row-major), C: MxN
//   f16 WMMA compute, fp32 accumulate.  act: 0 = none, 1 = exact GELU.
// Block: 256 threads = 8 wave32s; block tile 16(M) x 128(N); K chunked by 32,
// double-buffered LDS (one barrier per chunk), strength-reduced addressing
// (per-thread running pointers; no index multiplies in the steady state).
// ---------------------------------------------------------------------------
__global__ __launch_bounds__(256)
void gemm_wmma_f16(const float* __restrict__ A, int lda, long aBatch,
                   const float* __restrict__ B, int ldb, long bBatch,
                   const float* __restrict__ bias, int biasBatch,
                   float* __restrict__ C, int ldc, long cBatch,
                   int K, int act)
{
    __shared__ __attribute__((aligned(16))) _Float16 lA[2][16 * LDS_STRIDE];
    __shared__ __attribute__((aligned(16))) _Float16 lB[2][128 * LDS_STRIDE];

    const int t    = threadIdx.x;
    const int lane = t & 31;
    const int wave = t >> 5;
    const int m0   = blockIdx.y * 16;
    const int n0   = blockIdx.x * 128;
    const int eb   = blockIdx.z;

    const int mrow = lane & 15;   // M (A frag) / N (B frag) index within tile
    const int half = lane >> 4;   // lane half selects K sub-runs

    // ---- per-thread staging coordinates (all uniform, no guards) ----
    const int ar  = t >> 4;            // A row 0..15
    const int akc = (t & 15) * 2;      // A k-col pair base 0..30
    const int bn  = t & 127;           // B column (fixed per thread!)
    const int bk  = (t >> 7) * 2;      // B starting k row (0 or 2)

    // Running global pointers: A advances 32 floats / chunk, B 32 rows / chunk.
    const float* aPtr = A + (long)eb * aBatch + (long)(m0 + ar) * lda + akc;
    const float* bPtr = B + (long)eb * bBatch + (long)bk * ldb + n0 + bn;
    const long   bStepJ = 4L * ldb;    // rows advance 4 per j iteration
    const long   bStepC = 32L * ldb;   // rows advance 32 per chunk

    // Precomputed LDS half-indices
    const int laStore = ar * LDS_STRIDE + akc;      // A staging store
    const int lbStore = bn * LDS_STRIDE + bk;       // B staging store (+4j)
    const int laFrag  = mrow * LDS_STRIDE + half * 8;
    const int lbFrag  = (wave * 16 + mrow) * LDS_STRIDE + half * 8;

    // Stage one 32-wide K chunk into LDS buffer `buf`; advances pointers.
    auto stage = [&](int buf) {
        // A tile: 16 x 32, one float2 -> packed half2 per thread
        {
            float2 va = *(const float2*)aPtr;
            HPack p; p.h[0] = (_Float16)va.x; p.h[1] = (_Float16)va.y;
            *(unsigned int*)&lA[buf][laStore] = p.u;
        }
        // B tile: 32 x 128 transposed into LDS [n][k]; k-pairs packed as b32
        const float* p0 = bPtr;
        #pragma unroll
        for (int j = 0; j < 8; ++j) {
            float b0 = p0[0];
            float b1 = p0[ldb];
            HPack p; p.h[0] = (_Float16)b0; p.h[1] = (_Float16)b1;
            *(unsigned int*)&lB[buf][lbStore + 4 * j] = p.u;
            p0 += bStepJ;
        }
        aPtr += 32;
        bPtr += bStepC;
    };

    v8f acc = {};
    const int nChunks = K >> 5;

    stage(0);
    __syncthreads();

    for (int c = 0; c < nChunks; ++c) {
        const int cur = c & 1;
        if (c + 1 < nChunks) stage(cur ^ 1);

        // A fragment: lane half h holds K runs [8h,8h+8) and [16+8h, 16+8h+8)
        Frag16 af, bf;
        af.h[0] = *(const h8*)&lA[cur][laFrag];
        af.h[1] = *(const h8*)&lA[cur][laFrag + 16];
        // B fragment: lane half h holds K runs [8h,8h+8) and [16+8h, 16+8h+8) of column n
        bf.h[0] = *(const h8*)&lB[cur][lbFrag];
        bf.h[1] = *(const h8*)&lB[cur][lbFrag + 16];

        acc = __builtin_amdgcn_wmma_f32_16x16x32_f16(
                  false, af.v, false, bf.v, (short)0, acc, false, false);
        __syncthreads();   // one barrier per chunk (double-buffered)
    }

    // Epilogue: C/D layout — VGPR r, lane L: M = 8*(L/16)+r, N = L%16
    const float* bvec = bias + (long)eb * biasBatch;
    float* Cp = C + (long)eb * cBatch;
    const int col = n0 + wave * 16 + mrow;
    const float bb = bvec[col];
    const int mb = m0 + half * 8;
    #pragma unroll
    for (int r = 0; r < 8; ++r) {
        float x = acc[r] + bb;
        if (act) x = 0.5f * x * (1.0f + erff(x * 0.70710678118654752f));
        Cp[(long)(mb + r) * ldc + col] = x;
    }
}

// ---------------------------------------------------------------------------
// Gate: logits = fused @ gate_W + gate_b; softmax; top-2 (stable, renormed)
// One block of 256 threads: thread t -> (b = t/8, e = t%8)
// ---------------------------------------------------------------------------
__global__ __launch_bounds__(256)
void gate_kernel(const float* __restrict__ fused, const float* __restrict__ gW,
                 const float* __restrict__ gb, float* __restrict__ gate_probs,
                 float* __restrict__ topk_p, int* __restrict__ topk_i)
{
    __shared__ float sl[32][8];
    const int t = threadIdx.x;
    const int b = t >> 3, e = t & 7;
    float s = gb[e];
    const float* f = fused + b * 512;
    for (int d = 0; d < 512; ++d) s += f[d] * gW[d * 8 + e];
    sl[b][e] = s;
    __syncthreads();
    if (e == 0) {
        float mx = sl[b][0];
        for (int i = 1; i < 8; ++i) mx = fmaxf(mx, sl[b][i]);
        float p[8], den = 0.0f;
        for (int i = 0; i < 8; ++i) { p[i] = expf(sl[b][i] - mx); den += p[i]; }
        int   i0 = 0; float p0 = -1.0f;
        for (int i = 0; i < 8; ++i) {
            p[i] /= den;
            gate_probs[b * 8 + i] = p[i];
            if (p[i] > p0) { p0 = p[i]; i0 = i; }
        }
        int i1 = 0; float p1 = -1.0f;
        for (int i = 0; i < 8; ++i)
            if (i != i0 && p[i] > p1) { p1 = p[i]; i1 = i; }
        const float inv = 1.0f / (p0 + p1);
        topk_p[b * 2 + 0] = p0 * inv;
        topk_p[b * 2 + 1] = p1 * inv;
        topk_i[b * 2 + 0] = i0;
        topk_i[b * 2 + 1] = i1;
    }
}

// ---------------------------------------------------------------------------
// Aux loss: usage = mean_b gate_probs; aux = 0.1 * sum(u*log(1/8) - log(u)/8)
// ---------------------------------------------------------------------------
__global__ void aux_kernel(const float* __restrict__ gate_probs,
                           float* __restrict__ out_aux)
{
    __shared__ float su[8];
    const int e = threadIdx.x;
    if (e < 8) {
        float u = 0.0f;
        for (int b = 0; b < 32; ++b) u += gate_probs[b * 8 + e];
        u *= (1.0f / 32.0f);
        su[e] = u * logf(0.125f) - logf(u) * 0.125f;
    }
    __syncthreads();
    if (e == 0) {
        float s = 0.0f;
        for (int i = 0; i < 8; ++i) s += su[i];
        *out_aux = 0.1f * s;
    }
}

// ---------------------------------------------------------------------------
// Combine top-2 experts + LayerNorm + classifier.  One block per batch row.
// ---------------------------------------------------------------------------
__global__ __launch_bounds__(256)
void combine_kernel(const float* __restrict__ eo, const float* __restrict__ topk_p,
                    const int* __restrict__ topk_i, const float* __restrict__ ln_g,
                    const float* __restrict__ ln_b, const float* __restrict__ clsW,
                    const float* __restrict__ clsb, float* __restrict__ out)
{
    __shared__ float r0[256];
    __shared__ float r1[256];
    const int b = blockIdx.x, h = threadIdx.x;
    const int i0 = topk_i[b * 2], i1 = topk_i[b * 2 + 1];
    const float p0 = topk_p[b * 2], p1 = topk_p[b * 2 + 1];
    const float x = p0 * eo[((long)i0 * 32 + b) * 256 + h]
                  + p1 * eo[((long)i1 * 32 + b) * 256 + h];
    r0[h] = x; r1[h] = x * x;
    __syncthreads();
    for (int s = 128; s > 0; s >>= 1) {
        if (h < s) { r0[h] += r0[h + s]; r1[h] += r1[h + s]; }
        __syncthreads();
    }
    const float mu  = r0[0] * (1.0f / 256.0f);
    const float var = r1[0] * (1.0f / 256.0f) - mu * mu;
    const float xn  = (x - mu) * rsqrtf(var + 1e-5f);
    const float y   = xn * ln_g[h] + ln_b[h];
    __syncthreads();
    r0[h] = y * clsW[h * 2 + 0];
    r1[h] = y * clsW[h * 2 + 1];
    __syncthreads();
    for (int s = 128; s > 0; s >>= 1) {
        if (h < s) { r0[h] += r0[h + s]; r1[h] += r1[h + s]; }
        __syncthreads();
    }
    if (h == 0) {
        out[b * 2 + 0] = r0[0] + clsb[0];
        out[b * 2 + 1] = r1[0] + clsb[1];
    }
}

// ---------------------------------------------------------------------------
// Launch: attn weight == 1 exactly => EEG branch (wemb/pooled/eeg_emb/q/k) is
// dead code; logits/aux depend only on smiles path + MoE + classifier.
// ---------------------------------------------------------------------------
extern "C" void kernel_launch(void* const* d_in, const int* in_sizes, int n_in,
                              void* d_out, int out_size, void* d_ws, size_t ws_size,
                              hipStream_t stream)
{
    (void)in_sizes; (void)n_in; (void)out_size; (void)ws_size;

    const float* smiles_emb = (const float*)d_in[2];   // (32, 768)
    const float* smiles_W   = (const float*)d_in[7];   // (768, 256)
    const float* smiles_b   = (const float*)d_in[8];
    const float* Wv         = (const float*)d_in[13];  // (256, 256)
    const float* bv         = (const float*)d_in[14];
    const float* Wo         = (const float*)d_in[15];  // (256, 256)
    const float* bo         = (const float*)d_in[16];
    const float* gate_W     = (const float*)d_in[17];  // (512, 8)
    const float* gate_b     = (const float*)d_in[18];
    const float* exp_W1     = (const float*)d_in[19];  // (8, 512, 512)
    const float* exp_b1     = (const float*)d_in[20];  // (8, 512)
    const float* exp_W2     = (const float*)d_in[21];  // (8, 512, 256)
    const float* exp_b2     = (const float*)d_in[22];  // (8, 256)
    const float* ln_g       = (const float*)d_in[23];
    const float* ln_b       = (const float*)d_in[24];
    const float* cls_W      = (const float*)d_in[25];  // (256, 2)
    const float* cls_b      = (const float*)d_in[26];

    float* ws    = (float*)d_ws;
    float* fused = ws;               // 32 x 512  ([attended | smiles_proj])
    float* vbuf  = ws + 16384;       // 32 x 256
    float* h1    = ws + 24576;       // 8 x 32 x 512
    float* eo    = ws + 155648;      // 8 x 32 x 256
    float* gp    = ws + 221184;      // 32 x 8
    float* tp    = ws + 221440;      // 32 x 2
    int*   ti    = (int*)(ws + 221504); // 32 x 2

    float* out_logits = (float*)d_out;       // 64 floats
    float* out_aux    = (float*)d_out + 64;  // 1 float

    // 1) smiles_proj = smiles_emb @ smiles_W + smiles_b  -> fused[:, 256:512]
    gemm_wmma_f16<<<dim3(2, 2, 1), 256, 0, stream>>>(
        smiles_emb, 768, 0, smiles_W, 256, 0, smiles_b, 0,
        fused + 256, 512, 0, 768, 0);

    // 2) v = smiles_proj @ Wv + bv
    gemm_wmma_f16<<<dim3(2, 2, 1), 256, 0, stream>>>(
        fused + 256, 512, 0, Wv, 256, 0, bv, 0,
        vbuf, 256, 0, 256, 0);

    // 3) attended = v @ Wo + bo  -> fused[:, 0:256]   (attn weight == 1)
    gemm_wmma_f16<<<dim3(2, 2, 1), 256, 0, stream>>>(
        vbuf, 256, 0, Wo, 256, 0, bo, 0,
        fused, 512, 0, 256, 0);

    // 4) h1[e] = gelu(fused @ exp_W1[e] + exp_b1[e])   (batched over 8 experts)
    gemm_wmma_f16<<<dim3(4, 2, 8), 256, 0, stream>>>(
        fused, 512, 0, exp_W1, 512, 512L * 512, exp_b1, 512,
        h1, 512, 32L * 512, 512, 1);

    // 5) expert_out[e] = h1[e] @ exp_W2[e] + exp_b2[e]
    gemm_wmma_f16<<<dim3(2, 2, 8), 256, 0, stream>>>(
        h1, 512, 32L * 512, exp_W2, 256, 512L * 256, exp_b2, 256,
        eo, 256, 32L * 256, 512, 0);

    // 6) gate softmax + top-2
    gate_kernel<<<1, 256, 0, stream>>>(fused, gate_W, gate_b, gp, tp, ti);

    // 7) aux loss
    aux_kernel<<<1, 32, 0, stream>>>(gp, out_aux);

    // 8) combine experts + LayerNorm + classifier
    combine_kernel<<<32, 256, 0, stream>>>(eo, tp, ti, ln_g, ln_b, cls_W, cls_b,
                                           out_logits);
}
